// RelativeMultiHeadAttention_3049426780788
// MI455X (gfx1250) — compile-verified, matrix operand order checked by reading, not run on hardware
//
#include <hip/hip_runtime.h>
#include <hip/hip_bf16.h>

// Shapes (fixed by reference)
#define Bsz 4
#define Lq  2048
#define Dm  1024
#define Hh  16
#define DK  64

typedef __attribute__((ext_vector_type(16))) _Float16 v16h;
typedef __attribute__((ext_vector_type(8)))  float    v8f;
typedef __attribute__((ext_vector_type(4)))  unsigned int u32x4;

// ---------------------------------------------------------------------------
// Fragment loader: 16x32 f16 tile, "row per lane" layout (ISA 7.12.2):
//   lane l (<16): row M=l, holds K=0..7 and 16..23
//   lane l (>=16): row M=l-16, holds K=8..15 and 24..31
// Works for A fragments and (symmetric N-on-lanes layout) B fragments whose
// memory is [n][k] with k contiguous (K^T, W^T, V^T): two b128 loads.
// ---------------------------------------------------------------------------
__device__ __forceinline__ v16h load_frag16x32(const _Float16* __restrict__ src,
                                               int ld, int lane) {
  const int row = lane & 15;
  const int kb  = (lane >> 4) << 3;  // 0 or 8
  const _Float16* p = src + (size_t)row * ld + kb;
  union { v16h h; u32x4 u[2]; } f;
  f.u[0] = *(const u32x4*)(p);
  f.u[1] = *(const u32x4*)(p + 16);
  return f.h;
}

__device__ __forceinline__ v8f zero8() {
  v8f z;
#pragma unroll
  for (int i = 0; i < 8; ++i) z[i] = 0.0f;
  return z;
}

#define WMMA_F16(a, b, c) \
  __builtin_amdgcn_wmma_f32_16x16x32_f16(false, (a), false, (b), (short)0, (c), false, false)

// ---------------------------------------------------------------------------
// fp32 -> fp16 conversion kernels
// ---------------------------------------------------------------------------
__global__ void cvt_f32_f16(const float* __restrict__ s, _Float16* __restrict__ d, int n) {
  int i = blockIdx.x * blockDim.x + threadIdx.x;
  if (i < n) d[i] = (_Float16)s[i];
}

// w_rel: (2L-1, DK) -> padded to 2L rows (last row zero, touched by band tiles)
__global__ void cvt_wrel(const float* __restrict__ s, _Float16* __restrict__ d,
                         int nvalid, int ntotal) {
  int i = blockIdx.x * blockDim.x + threadIdx.x;
  if (i < ntotal) d[i] = (i < nvalid) ? (_Float16)s[i] : (_Float16)0.0f;
}

// ---------------------------------------------------------------------------
// Projection GEMM:  out[m,n] = scale * ( sum_k X[m,k]*W[n,k] + bias[n] )
// One wave computes a 32x64 tile. Ping-pong pipelining (unroll-by-2, two
// fragment register sets alternating roles): no cross-set register copies,
// refill loads for set S issue between set-S WMMAs and set-(1-S) WMMAs, so
// each load batch gets an 8-WMMA latency window.
// mode 0: store f16 as (B,H,L,DK)      (Q with scale=1/8, K with scale=1)
// mode 1: store f16 as (B,H,DK,L)      (V transposed for flash B-fragments)
// mode 2: store f32 row-major (M,N)    (final output projection)
// ---------------------------------------------------------------------------
__launch_bounds__(32)
__global__ void proj_wmma(const _Float16* __restrict__ X, const _Float16* __restrict__ W,
                          const float* __restrict__ bias, _Float16* __restrict__ out16,
                          float* __restrict__ out32, int mode, float scale) {
  const int lane = threadIdx.x & 31;
  const int hlf  = lane >> 4;
  const int nl   = lane & 15;
  const int m0   = blockIdx.x * 32;
  const int n0   = blockIdx.y * 64;

  const _Float16* Xp = X + (size_t)m0 * Dm;
  const _Float16* Wp = W + (size_t)n0 * Dm;

  v8f acc[2][4];
#pragma unroll
  for (int cf = 0; cf < 4; ++cf) {
    float bv = bias[n0 + cf * 16 + nl];   // whole acc column shares N
#pragma unroll
    for (int r = 0; r < 2; ++r)
#pragma unroll
      for (int i = 0; i < 8; ++i) acc[r][cf][i] = bv;
  }

  // prologue: set0 <- chunk 0, set1 <- chunk 32
  v16h a0[2], b0f[4], a1[2], b1f[4];
#pragma unroll
  for (int r = 0; r < 2; ++r) a0[r] = load_frag16x32(Xp + (size_t)r * 16 * Dm, Dm, lane);
#pragma unroll
  for (int cf = 0; cf < 4; ++cf) b0f[cf] = load_frag16x32(Wp + (size_t)cf * 16 * Dm, Dm, lane);
#pragma unroll
  for (int r = 0; r < 2; ++r) a1[r] = load_frag16x32(Xp + (size_t)r * 16 * Dm + 32, Dm, lane);
#pragma unroll
  for (int cf = 0; cf < 4; ++cf) b1f[cf] = load_frag16x32(Wp + (size_t)cf * 16 * Dm + 32, Dm, lane);

  for (int kc = 0; kc < Dm; kc += 64) {
    __builtin_prefetch(Xp + (size_t)lane * Dm + kc + 128, 0, 3);      // global_prefetch_b8
    __builtin_prefetch(Wp + (size_t)(lane * 2) * Dm + kc + 128, 0, 3);

    // compute chunk kc with set0
#pragma unroll
    for (int r = 0; r < 2; ++r)
#pragma unroll
      for (int cf = 0; cf < 4; ++cf)
        acc[r][cf] = WMMA_F16(a0[r], b0f[cf], acc[r][cf]);

    // refill set0 <- chunk kc+64 (overlaps set1 WMMAs below)
    const int k2 = (kc + 64 < Dm) ? kc + 64 : kc;
#pragma unroll
    for (int r = 0; r < 2; ++r) a0[r] = load_frag16x32(Xp + (size_t)r * 16 * Dm + k2, Dm, lane);
#pragma unroll
    for (int cf = 0; cf < 4; ++cf) b0f[cf] = load_frag16x32(Wp + (size_t)cf * 16 * Dm + k2, Dm, lane);

    // compute chunk kc+32 with set1
#pragma unroll
    for (int r = 0; r < 2; ++r)
#pragma unroll
      for (int cf = 0; cf < 4; ++cf)
        acc[r][cf] = WMMA_F16(a1[r], b1f[cf], acc[r][cf]);

    // refill set1 <- chunk kc+96 (overlaps next iteration's set0 WMMAs)
    const int k3 = (kc + 96 < Dm) ? kc + 96 : kc;
#pragma unroll
    for (int r = 0; r < 2; ++r) a1[r] = load_frag16x32(Xp + (size_t)r * 16 * Dm + k3, Dm, lane);
#pragma unroll
    for (int cf = 0; cf < 4; ++cf) b1f[cf] = load_frag16x32(Wp + (size_t)cf * 16 * Dm + k3, Dm, lane);
  }

#pragma unroll
  for (int r = 0; r < 2; ++r)
#pragma unroll
    for (int cf = 0; cf < 4; ++cf)
#pragma unroll
      for (int i = 0; i < 8; ++i) {
        const int m = m0 + r * 16 + i + 8 * hlf;  // D-frag: M = i + 8*(lane>=16)
        const int n = n0 + cf * 16 + nl;          //         N = lane&15
        const float v = acc[r][cf][i] * scale;
        if (mode == 2) {
          out32[(size_t)m * Dm + n] = v;
        } else {
          const int b = m >> 11, l = m & (Lq - 1);
          const int h = n >> 6,  dk = n & (DK - 1);
          if (mode == 0)
            out16[(((size_t)(b * Hh + h) * Lq + l) * DK) + dk] = (_Float16)v;
          else
            out16[(((size_t)(b * Hh + h) * DK + dk) * Lq) + l] = (_Float16)v;
        }
      }
}

// ---------------------------------------------------------------------------
// Flash attention with relative bias.
// One wave handles 16 query rows of one (b,h); loops 32 keys per step.
//   S  = Qs * K^T                      (Qs already scaled by 1/sqrt(dk))
//   T  = Qs * Wrel^T over 48-wide diagonal window; bias[r,c] = T[r][r+c]
//   online softmax, O += P * V
// V fragments are loaded at the TOP of each iteration so their latency is
// hidden behind the score/bias WMMAs and the softmax VALU work.
// ---------------------------------------------------------------------------
__launch_bounds__(32)
__global__ void flash_attn_rel(const _Float16* __restrict__ Qh, const _Float16* __restrict__ Kh,
                               const _Float16* __restrict__ Vt, const _Float16* __restrict__ Wr,
                               _Float16* __restrict__ Ocat) {
  __shared__ __align__(16) float    Tlds[16][48];
  __shared__ __align__(16) _Float16 Plds[16][32];

  const int lane = threadIdx.x & 31;
  const int hlf  = lane >> 4;
  const int nl   = lane & 15;
  const int bh = blockIdx.y;
  const int b = bh >> 4, h = bh & 15;
  const int r0 = blockIdx.x * 16;

  const size_t qbase = ((size_t)bh * Lq + r0) * DK;
  const v16h aq0 = load_frag16x32(Qh + qbase,      DK, lane);   // d = 0..31
  const v16h aq1 = load_frag16x32(Qh + qbase + 32, DK, lane);   // d = 32..63

  float mrow[8], lrow[8];
  v8f o[4];
#pragma unroll
  for (int i = 0; i < 8; ++i) { mrow[i] = -3.0e38f; lrow[i] = 0.0f; }
#pragma unroll
  for (int cf = 0; cf < 4; ++cf) o[cf] = zero8();

  for (int c0 = 0; c0 < Lq; c0 += 32) {
    // V fragments for THIS block, issued early (used after softmax)
    v16h vb[4];
#pragma unroll
    for (int cf = 0; cf < 4; ++cf)
      vb[cf] = load_frag16x32(Vt + ((size_t)bh * DK + cf * 16) * Lq + c0, Lq, lane);

    // warm next block's K / V / w_rel rows (global_prefetch_b8, lanes spread rows)
    if (c0 + 32 < Lq) {
      __builtin_prefetch(Kh + ((size_t)bh * Lq + c0 + 32 + lane) * DK, 0, 3);
      __builtin_prefetch(Vt + ((size_t)bh * DK + lane * 2) * Lq + c0 + 32, 0, 3);
      __builtin_prefetch(Wr + (size_t)(r0 + c0 + 48 + lane) * DK, 0, 3);
    }

    // ---- content scores: two 16x16 subtiles over this 32-key block
    v8f s[2];
#pragma unroll
    for (int st = 0; st < 2; ++st) {
      const _Float16* kp = Kh + ((size_t)bh * Lq + c0 + st * 16) * DK;
      v16h b0 = load_frag16x32(kp,      DK, lane);
      v16h b1 = load_frag16x32(kp + 32, DK, lane);
      v8f acc = zero8();
      acc = WMMA_F16(aq0, b0, acc);
      acc = WMMA_F16(aq1, b1, acc);
      s[st] = acc;
    }

    // ---- relative-bias band: T[r][m] = Qs[r] . Wrel[r0+c0+m], m in [0,48)
    const int sbase = r0 + c0;
#pragma unroll
    for (int mt = 0; mt < 3; ++mt) {
      const _Float16* wp = Wr + (size_t)(sbase + mt * 16) * DK;
      v16h b0 = load_frag16x32(wp,      DK, lane);
      v16h b1 = load_frag16x32(wp + 32, DK, lane);
      v8f acc = zero8();
      acc = WMMA_F16(aq0, b0, acc);
      acc = WMMA_F16(aq1, b1, acc);
#pragma unroll
      for (int i = 0; i < 8; ++i) Tlds[i + 8 * hlf][mt * 16 + nl] = acc[i];
    }
    __syncthreads();

    // ---- skew gather: bias[r,c] = T[r][r+c]  (consecutive lanes -> no conflicts)
#pragma unroll
    for (int st = 0; st < 2; ++st)
#pragma unroll
      for (int i = 0; i < 8; ++i) {
        const int M = i + 8 * hlf;
        s[st][i] += Tlds[M][M + st * 16 + nl];
      }

    // ---- online softmax over the 32-key block
#pragma unroll
    for (int i = 0; i < 8; ++i) {
      float v = fmaxf(s[0][i], s[1][i]);
#pragma unroll
      for (int off = 1; off < 16; off <<= 1) v = fmaxf(v, __shfl_xor(v, off, 32));
      const float mnew = fmaxf(mrow[i], v);
      const float corr = __expf(mrow[i] - mnew);
      const float p0 = __expf(s[0][i] - mnew);
      const float p1 = __expf(s[1][i] - mnew);
      float rs = p0 + p1;
#pragma unroll
      for (int off = 1; off < 16; off <<= 1) rs += __shfl_xor(rs, off, 32);
      lrow[i] = lrow[i] * corr + rs;
      mrow[i] = mnew;
#pragma unroll
      for (int cf = 0; cf < 4; ++cf) o[cf][i] *= corr;
      Plds[i + 8 * hlf][nl]      = (_Float16)p0;
      Plds[i + 8 * hlf][16 + nl] = (_Float16)p1;
    }
    __syncthreads();

    // ---- O += P(16x32) * V(32x64): P re-read in A layout from LDS
    const v16h pa = load_frag16x32(&Plds[0][0], 32, lane);
#pragma unroll
    for (int cf = 0; cf < 4; ++cf)
      o[cf] = WMMA_F16(pa, vb[cf], o[cf]);
    __syncthreads();
  }

  // ---- finalize + store to concat layout (B, L, H*DK) f16
#pragma unroll
  for (int cf = 0; cf < 4; ++cf)
#pragma unroll
    for (int i = 0; i < 8; ++i) {
      const int M = i + 8 * hlf;
      const float v = o[cf][i] / lrow[i];
      const size_t idx = ((size_t)b * Lq + r0 + M) * Dm + h * DK + cf * 16 + nl;
      Ocat[idx] = (_Float16)v;
    }
}

// ---------------------------------------------------------------------------
extern "C" void kernel_launch(void* const* d_in, const int* in_sizes, int n_in,
                              void* d_out, int out_size, void* d_ws, size_t ws_size,
                              hipStream_t stream) {
  (void)in_sizes; (void)n_in; (void)out_size; (void)ws_size;
  const float* q    = (const float*)d_in[0];
  const float* k    = (const float*)d_in[1];
  const float* v    = (const float*)d_in[2];
  const float* Wq   = (const float*)d_in[3];
  const float* bq   = (const float*)d_in[4];
  const float* Wk   = (const float*)d_in[5];
  const float* bk   = (const float*)d_in[6];
  const float* Wv   = (const float*)d_in[7];
  const float* bv   = (const float*)d_in[8];
  const float* Wo   = (const float*)d_in[9];
  const float* bo   = (const float*)d_in[10];
  const float* wrel = (const float*)d_in[11];
  float* out = (float*)d_out;

  char* ws = (char*)d_ws;
  size_t off = 0;
  auto alloc = [&](size_t bytes) -> void* {
    void* p = ws + off;
    off = (off + bytes + 255) & ~(size_t)255;
    return p;
  };

  const size_t nX = (size_t)Bsz * Lq * Dm;   // 8,388,608
  const size_t nW = (size_t)Dm * Dm;         // 1,048,576
  const size_t nRv = (size_t)(2 * Lq - 1) * DK;
  const size_t nRt = (size_t)(2 * Lq) * DK;  // padded

  _Float16* Xq16 = (_Float16*)alloc(nX * 2);
  _Float16* Xk16 = (_Float16*)alloc(nX * 2);
  _Float16* Xv16 = (_Float16*)alloc(nX * 2);
  _Float16* Wq16 = (_Float16*)alloc(nW * 2);
  _Float16* Wk16 = (_Float16*)alloc(nW * 2);
  _Float16* Wv16 = (_Float16*)alloc(nW * 2);
  _Float16* Wo16 = (_Float16*)alloc(nW * 2);
  _Float16* Wr16 = (_Float16*)alloc(nRt * 2);
  _Float16* Q16  = (_Float16*)alloc(nX * 2);  // (B,H,L,DK), pre-scaled 1/8
  _Float16* K16  = (_Float16*)alloc(nX * 2);  // (B,H,L,DK)
  _Float16* VT16 = (_Float16*)alloc(nX * 2);  // (B,H,DK,L)
  _Float16* Oc16 = (_Float16*)alloc(nX * 2);  // (B,L,D) concat

  const int TPB = 256;
  cvt_f32_f16<<<dim3((nX + TPB - 1) / TPB), dim3(TPB), 0, stream>>>(q, Xq16, (int)nX);
  cvt_f32_f16<<<dim3((nX + TPB - 1) / TPB), dim3(TPB), 0, stream>>>(k, Xk16, (int)nX);
  cvt_f32_f16<<<dim3((nX + TPB - 1) / TPB), dim3(TPB), 0, stream>>>(v, Xv16, (int)nX);
  cvt_f32_f16<<<dim3((nW + TPB - 1) / TPB), dim3(TPB), 0, stream>>>(Wq, Wq16, (int)nW);
  cvt_f32_f16<<<dim3((nW + TPB - 1) / TPB), dim3(TPB), 0, stream>>>(Wk, Wk16, (int)nW);
  cvt_f32_f16<<<dim3((nW + TPB - 1) / TPB), dim3(TPB), 0, stream>>>(Wv, Wv16, (int)nW);
  cvt_f32_f16<<<dim3((nW + TPB - 1) / TPB), dim3(TPB), 0, stream>>>(Wo, Wo16, (int)nW);
  cvt_wrel<<<dim3((nRt + TPB - 1) / TPB), dim3(TPB), 0, stream>>>(wrel, Wr16, (int)nRv, (int)nRt);

  // Projections: M = B*L = 8192 rows, N = 1024 cols, wave tile 32x64
  const dim3 pgrid(Bsz * Lq / 32, Dm / 64), pblk(32);
  const float inv_sqrt_dk = 0.125f;  // 1/sqrt(64)
  proj_wmma<<<pgrid, pblk, 0, stream>>>(Xq16, Wq16, bq, Q16,  nullptr, 0, inv_sqrt_dk);
  proj_wmma<<<pgrid, pblk, 0, stream>>>(Xk16, Wk16, bk, K16,  nullptr, 0, 1.0f);
  proj_wmma<<<pgrid, pblk, 0, stream>>>(Xv16, Wv16, bv, VT16, nullptr, 1, 1.0f);

  // Flash attention: one wave per 16 query rows per (b,h)
  flash_attn_rel<<<dim3(Lq / 16, Bsz * Hh), dim3(32), 0, stream>>>(Q16, K16, VT16, Wr16, Oc16);

  // Output projection -> f32
  proj_wmma<<<pgrid, pblk, 0, stream>>>(Oc16, Wo16, bo, nullptr, out, 2, 1.0f);
}
